// AutoCorrelationAttention_76785425317988
// MI455X (gfx1250) — matrix operand order repaired and checked by reading, not compiled, not run on hardware
//
#include <hip/hip_runtime.h>
#include <hip/hip_bf16.h>

#define BATCH 16
#define SEQ   2048
#define DMODEL 512

typedef __attribute__((ext_vector_type(16))) __bf16 v16bf;
typedef __attribute__((ext_vector_type(8)))  float  v8f;

static __device__ inline v8f wmma_bf16(v16bf a, v16bf b, v8f c) {
  return __builtin_amdgcn_wmma_f32_16x16x32_bf16(false, a, false, b, (short)0, c, false, false);
}

// A fragment (16x32 bf16, MxK): lane<16 -> row=lane, K = {k0..k0+7, k0+16..k0+23}
//                               lane>=16 -> row=lane-16, K = {k0+8..15, k0+24..31}
static __device__ inline v16bf load_frag_a(const __bf16* base, int ld, int m0, int k0, int lane) {
  const int h = lane >> 4, r = lane & 15;
  const __bf16* p = base + (size_t)(m0 + r) * ld + k0 + h * 8;
  union { uint4 q; __bf16 e[8]; } u0, u1;
  u0.q = *(const uint4*)(p);
  u1.q = *(const uint4*)(p + 16);
  v16bf f;
#pragma unroll
  for (int i = 0; i < 8; ++i) { f[i] = u0.e[i]; f[8 + i] = u1.e[i]; }
  return f;
}

// Same A fragment but source is f32, converted to bf16 on the fly.
static __device__ inline v16bf load_frag_a_f32(const float* base, int ld, int m0, int k0, int lane) {
  const int h = lane >> 4, r = lane & 15;
  const float* p = base + (size_t)(m0 + r) * ld + k0 + h * 8;
  v16bf f;
#pragma unroll
  for (int i = 0; i < 8; ++i) { f[i] = (__bf16)p[i]; f[8 + i] = (__bf16)p[16 + i]; }
  return f;
}

// B fragment (32x16 bf16, KxN): lane holds column n = lane&15;
// lanes 0-15: K = {k0..k0+7, k0+16..k0+23}, lanes 16-31: K = {k0+8..15, k0+24..31} (in the
// memory row that holds column n's K-data; row stride = ld).
static __device__ inline v16bf load_frag_b(const __bf16* base, int ld, int n0, int k0, int lane) {
  const int h = lane >> 4, n = lane & 15;
  const __bf16* p = base + (size_t)(n0 + n) * ld + k0 + h * 8;
  union { uint4 q; __bf16 e[8]; } u0, u1;
  u0.q = *(const uint4*)(p);
  u1.q = *(const uint4*)(p + 16);
  v16bf f;
#pragma unroll
  for (int i = 0; i < 8; ++i) { f[i] = u0.e[i]; f[8 + i] = u1.e[i]; }
  return f;
}

// ---------------------------------------------------------------- weights f32 -> bf16
__global__ __launch_bounds__(256) void cvt_weights_kernel(
    const float* __restrict__ Wq, const float* __restrict__ Wk,
    const float* __restrict__ Wv, const float* __restrict__ Wo,
    __bf16* __restrict__ dst) {
  const int i = blockIdx.x * 256 + threadIdx.x;   // 4 * 512*512 total
  const int which = i >> 18;
  const int off = i & 262143;
  const float* s = (which == 0) ? Wq : (which == 1) ? Wk : (which == 2) ? Wv : Wo;
  dst[i] = (__bf16)s[off];
}

// ---------------------------------------------------------------- fused QKV projection
// Y = x @ W^T + b for W in {Wq,Wk,Wv}. A = x[M=32768, K=512] (f32, converted),
// B column n = W row n (contiguous K). Q,K stored [b][s][d]; V stored transposed [b][d][s].
__global__ __launch_bounds__(256) void qkv_proj_kernel(
    const float* __restrict__ x,
    const __bf16* __restrict__ Wq, const __bf16* __restrict__ Wk, const __bf16* __restrict__ Wv,
    const float* __restrict__ bq, const float* __restrict__ bk, const float* __restrict__ bv,
    __bf16* __restrict__ Q, __bf16* __restrict__ K, __bf16* __restrict__ Vt) {
  const int tid = threadIdx.x, lane = tid & 31, w = tid >> 5;
  const int m0 = blockIdx.x * 64 + (w & 3) * 16;   // 512 blocks in M
  const int n0 = blockIdx.y * 128 + (w >> 2) * 64; // 4 blocks in N

  v8f accq[4], acck[4], accv[4];
#pragma unroll
  for (int t = 0; t < 4; ++t) {
    accq[t] = (v8f){0.f,0.f,0.f,0.f,0.f,0.f,0.f,0.f};
    acck[t] = accq[t]; accv[t] = accq[t];
  }

  for (int k0 = 0; k0 < DMODEL; k0 += 32) {
    // Issue the A-fragment conversion and ALL 12 B-fragment loads first so the
    // VMEM pipe stays full; the 12 WMMAs then drain with partial loadcnt waits.
    v16bf a = load_frag_a_f32(x, DMODEL, m0, k0, lane);
    v16bf fq[4], fk[4], fv[4];
#pragma unroll
    for (int t = 0; t < 4; ++t) fq[t] = load_frag_b(Wq, DMODEL, n0 + t * 16, k0, lane);
#pragma unroll
    for (int t = 0; t < 4; ++t) fk[t] = load_frag_b(Wk, DMODEL, n0 + t * 16, k0, lane);
#pragma unroll
    for (int t = 0; t < 4; ++t) fv[t] = load_frag_b(Wv, DMODEL, n0 + t * 16, k0, lane);
#pragma unroll
    for (int t = 0; t < 4; ++t) accq[t] = wmma_bf16(a, fq[t], accq[t]);
#pragma unroll
    for (int t = 0; t < 4; ++t) acck[t] = wmma_bf16(a, fk[t], acck[t]);
#pragma unroll
    for (int t = 0; t < 4; ++t) accv[t] = wmma_bf16(a, fv[t], accv[t]);
  }

  const int h = lane >> 4, col = lane & 15;
#pragma unroll
  for (int t = 0; t < 4; ++t) {
    const int n = n0 + t * 16 + col;
    const float biq = bq[n], bik = bk[n], biv = bv[n];
#pragma unroll
    for (int r = 0; r < 8; ++r) {
      const int m = m0 + r + 8 * h;           // flattened (b,s)
      Q[(size_t)m * DMODEL + n] = (__bf16)(accq[t][r] + biq);
      K[(size_t)m * DMODEL + n] = (__bf16)(acck[t][r] + bik);
      const int b = m >> 11, s = m & 2047;    // transpose V: [b][d][s]
      Vt[((size_t)b * DMODEL + n) * SEQ + s] = (__bf16)(accv[t][r] + biv);
    }
  }
}

// ---------------------------------------------------------------- flash attention
// One block = (batch, 16-query tile). 8 waves split D into 64-col slices.
__global__ __launch_bounds__(256) void flash_attn_kernel(
    const __bf16* __restrict__ Q, const __bf16* __restrict__ K,
    const __bf16* __restrict__ Vt, __bf16* __restrict__ O) {
  __shared__ __align__(16) float  sS[8][16][32];   // per-wave partial scores
  __shared__ __align__(16) float  sRed[16][32];    // reduced scores
  __shared__ __align__(16) __bf16 sP[16][32];      // softmax probabilities (bf16)
  __shared__ float sM[16], sL[16], sAlpha[16];

  const int tid = threadIdx.x, lane = tid & 31, w = tid >> 5;
  const int b = blockIdx.x >> 7;        // 128 q-tiles per batch
  const int q0 = (blockIdx.x & 127) * 16;

  const __bf16* Qb = Q + (size_t)b * SEQ * DMODEL;
  const __bf16* Kb = K + (size_t)b * SEQ * DMODEL;
  const __bf16* Vb = Vt + (size_t)b * DMODEL * SEQ;

  // Q fragments for this wave's 64-wide D slice are loop-invariant: preload.
  v16bf qf0 = load_frag_a(Qb, DMODEL, q0, w * 64, lane);
  v16bf qf1 = load_frag_a(Qb, DMODEL, q0, w * 64 + 32, lane);

  v8f oacc[4];
#pragma unroll
  for (int t = 0; t < 4; ++t) oacc[t] = (v8f){0.f,0.f,0.f,0.f,0.f,0.f,0.f,0.f};

  if (tid < 16) { sM[tid] = -1.0e30f; sL[tid] = 0.f; }
  __syncthreads();

  const float scale = 0.044194173824159216f;  // 1/sqrt(512)
  const int h = lane >> 4, col = lane & 15;

  for (int kb = 0; kb < SEQ / 32; ++kb) {
    const int key0 = kb * 32;

    // Prefetch next key block (K rows + V^T columns) toward the WGP while this
    // block computes: emits global_prefetch_b8, one cacheline probe per lane.
    if (kb + 1 < SEQ / 32) {
      const int nk = key0 + 32;
      __builtin_prefetch(Kb + (size_t)(nk + lane) * DMODEL + w * 64, 0, 0);
      __builtin_prefetch(Vb + (size_t)(w * 64 + (lane & 15)) * SEQ + nk, 0, 0);
    }

    // --- partial scores over this wave's D-slice: S_w = Q_w * K_w^T  (16x32)
    {
      v16bf f00 = load_frag_b(Kb, DMODEL, key0,      w * 64,      lane);
      v16bf f01 = load_frag_b(Kb, DMODEL, key0,      w * 64 + 32, lane);
      v16bf f10 = load_frag_b(Kb, DMODEL, key0 + 16, w * 64,      lane);
      v16bf f11 = load_frag_b(Kb, DMODEL, key0 + 16, w * 64 + 32, lane);
      v8f s0 = (v8f){0.f,0.f,0.f,0.f,0.f,0.f,0.f,0.f};
      v8f s1 = s0;
      s0 = wmma_bf16(qf0, f00, s0);
      s1 = wmma_bf16(qf0, f10, s1);
      s0 = wmma_bf16(qf1, f01, s0);
      s1 = wmma_bf16(qf1, f11, s1);
#pragma unroll
      for (int r = 0; r < 8; ++r) {
        sS[w][r + 8 * h][col]      = s0[r];
        sS[w][r + 8 * h][16 + col] = s1[r];
      }
    }
    __syncthreads();

    // --- cross-wave reduction: 512 elems, 2 per thread
#pragma unroll
    for (int rep = 0; rep < 2; ++rep) {
      const int e = tid + rep * 256;
      const int m = e >> 5, n = e & 31;
      float s = 0.f;
#pragma unroll
      for (int ww = 0; ww < 8; ++ww) s += sS[ww][m][n];
      sRed[m][n] = s * scale;
    }
    __syncthreads();

    // --- online softmax row update (one row per thread 0..15)
    if (tid < 16) {
      const int m = tid;
      const float mOld = sM[m];
      float rmax = mOld;
      for (int n = 0; n < 32; ++n) rmax = fmaxf(rmax, sRed[m][n]);
      const float alpha = __expf(mOld - rmax);
      float lsum = 0.f;
      for (int n = 0; n < 32; ++n) {
        const float p = __expf(sRed[m][n] - rmax);
        lsum += p;
        sP[m][n] = (__bf16)p;
      }
      sM[m] = rmax;
      sL[m] = sL[m] * alpha + lsum;
      sAlpha[m] = alpha;
    }
    __syncthreads();

    // --- O_w = alpha * O_w + P * V_w   (V^T rows are contiguous in key index)
    {
      v16bf pf = load_frag_a(&sP[0][0], 32, 0, 0, lane);
      v16bf vf[4];
#pragma unroll
      for (int t = 0; t < 4; ++t)
        vf[t] = load_frag_b(Vb, SEQ, w * 64 + t * 16, key0, lane);
      float al[8];
#pragma unroll
      for (int r = 0; r < 8; ++r) al[r] = sAlpha[r + 8 * h];
#pragma unroll
      for (int t = 0; t < 4; ++t) {
        v8f a = oacc[t];
#pragma unroll
        for (int r = 0; r < 8; ++r) a[r] *= al[r];
        oacc[t] = wmma_bf16(pf, vf[t], a);
      }
    }
    __syncthreads();
  }

  // --- epilogue: normalize by running sum, write bf16
  float linv[8];
#pragma unroll
  for (int r = 0; r < 8; ++r) linv[r] = 1.0f / sL[r + 8 * h];
  __bf16* Ob = O + (size_t)b * SEQ * DMODEL;
#pragma unroll
  for (int t = 0; t < 4; ++t)
#pragma unroll
    for (int r = 0; r < 8; ++r)
      Ob[(size_t)(q0 + r + 8 * h) * DMODEL + w * 64 + t * 16 + col] =
          (__bf16)(oacc[t][r] * linv[r]);
}

// ---------------------------------------------------------------- output projection (f32 out)
__global__ __launch_bounds__(256) void out_proj_kernel(
    const __bf16* __restrict__ A, const __bf16* __restrict__ Wo,
    const float* __restrict__ bo, float* __restrict__ out) {
  const int tid = threadIdx.x, lane = tid & 31, w = tid >> 5;
  const int m0 = blockIdx.x * 64 + (w & 3) * 16;
  const int n0 = blockIdx.y * 128 + (w >> 2) * 64;

  v8f acc[4];
#pragma unroll
  for (int t = 0; t < 4; ++t) acc[t] = (v8f){0.f,0.f,0.f,0.f,0.f,0.f,0.f,0.f};

  for (int k0 = 0; k0 < DMODEL; k0 += 32) {
    v16bf a = load_frag_a(A, DMODEL, m0, k0, lane);
    v16bf f[4];
#pragma unroll
    for (int t = 0; t < 4; ++t) f[t] = load_frag_b(Wo, DMODEL, n0 + t * 16, k0, lane);
#pragma unroll
    for (int t = 0; t < 4; ++t) acc[t] = wmma_bf16(a, f[t], acc[t]);
  }

  const int h = lane >> 4, col = lane & 15;
#pragma unroll
  for (int t = 0; t < 4; ++t) {
    const int n = n0 + t * 16 + col;
    const float bi = bo[n];
#pragma unroll
    for (int r = 0; r < 8; ++r)
      out[(size_t)(m0 + r + 8 * h) * DMODEL + n] = acc[t][r] + bi;
  }
}

extern "C" void kernel_launch(void* const* d_in, const int* in_sizes, int n_in,
                              void* d_out, int out_size, void* d_ws, size_t ws_size,
                              hipStream_t stream) {
  const float* x  = (const float*)d_in[0];
  const float* Wq = (const float*)d_in[1];
  const float* bq = (const float*)d_in[2];
  const float* Wk = (const float*)d_in[3];
  const float* bk = (const float*)d_in[4];
  const float* Wv = (const float*)d_in[5];
  const float* bv = (const float*)d_in[6];
  const float* Wo = (const float*)d_in[7];
  const float* bo = (const float*)d_in[8];

  const size_t WELEM = (size_t)DMODEL * DMODEL;         // 262144
  const size_t TELEM = (size_t)BATCH * SEQ * DMODEL;    // 16777216

  __bf16* Wqb = (__bf16*)d_ws;
  __bf16* Wkb = Wqb + WELEM;
  __bf16* Wvb = Wkb + WELEM;
  __bf16* Wob = Wvb + WELEM;
  __bf16* Qb  = Wob + WELEM;
  __bf16* Kb  = Qb + TELEM;
  __bf16* Vt  = Kb + TELEM;
  __bf16* Ob  = Vt + TELEM;
  // total ws use: 2 MB + 4 * 32 MB = 130 MB

  cvt_weights_kernel<<<dim3((4 * WELEM) / 256), 256, 0, stream>>>(Wq, Wk, Wv, Wo, Wqb);
  qkv_proj_kernel<<<dim3(512, 4), 256, 0, stream>>>(x, Wqb, Wkb, Wvb, bq, bk, bv, Qb, Kb, Vt);
  flash_attn_kernel<<<dim3(BATCH * SEQ / 16), 256, 0, stream>>>(Qb, Kb, Vt, Ob);
  out_proj_kernel<<<dim3(512, 4), 256, 0, stream>>>(Ob, Wob, bo, (float*)d_out);
}
